// AGCRN_Model_37529424233023
// MI455X (gfx1250) — matrix operand order repaired and checked by reading, not compiled, not run on hardware
//
#include <hip/hip_runtime.h>
#include <math.h>

#define N_NODES 10000
#define N_EDGES 100000
#define T_STEPS 12
#define C_IN    16
#define C_EDGE  8
#define HID_C   32

typedef __attribute__((ext_vector_type(16))) _Float16 v16h;
typedef __attribute__((ext_vector_type(8)))  float    v8f;

union AFrag { v16h v; _Float16 e[16]; };
union CFrag { v8f  v; float    e[8];  };

// ---------------------------------------------------------------------------
// WMMA GEMM: C[M, COUT] = A[M, K] @ W[K, COUT]   (row-major, lda == K)
// K and COUT are compile-time so all fragment padding / strides fold to
// immediates. Each wave owns one 16-wide column strip and sweeps 4 M-tiles,
// reusing the B (weight) fragment -> 4 unrolled v_wmma per wave.
// ---------------------------------------------------------------------------
template<int K, int COUT>
__global__ void wmma_gemm(const float* __restrict__ A,
                          const float* __restrict__ W,
                          float* __restrict__ C, int M)
{
    constexpr int TILES_N = COUT / 16;
    int wave = blockIdx.x * (blockDim.x >> 5) + (threadIdx.x >> 5);
    int tn   = wave % TILES_N;
    int tmg  = wave / TILES_N;        // group of 4 m-tiles
    int n0   = tn << 4;

    int l  = threadIdx.x & 31;
    int g  = l >> 4;                  // half-wave select
    int mn = l & 15;                  // row (A) / col (B,C,D)

    // B fragment: 32x16 f16, lane = column n0+mn, K(h) = (h&8)*2 + (h&7) + g*8.
    // Loaded once, reused for all 4 M-tiles.
    AFrag b;
    const float* wcol = W + n0 + mn;
#pragma unroll
    for (int h = 0; h < 16; ++h) {
        int kb = ((h & 8) << 1) + (h & 7) + (g << 3);
        b.e[h] = (kb < K) ? (_Float16)wcol[kb * COUT] : (_Float16)0.0f;
    }

#pragma unroll
    for (int i = 0; i < 4; ++i) {
        int m0 = (tmg * 4 + i) << 4;
        if (m0 < M) {                 // uniform per-wave guard
            // A fragment: 16x32 f16, lane = row m0+mn,
            // K(h) = (h&8)*2 + (h&7) + g*8  (ISA 7.12.2)
            AFrag a;
            const float* arow = A + (size_t)(m0 + mn) * K;
#pragma unroll
            for (int h = 0; h < 16; ++h) {
                int ka = ((h & 8) << 1) + (h & 7) + (g << 3);
                a.e[h] = (ka < K) ? (_Float16)arow[ka] : (_Float16)0.0f;
            }

            CFrag c;
            c.v = (v8f){0.f, 0.f, 0.f, 0.f, 0.f, 0.f, 0.f, 0.f};
            c.v = __builtin_amdgcn_wmma_f32_16x16x32_f16(
                      false, a.v, false, b.v, (short)0, c.v, false, false);

            // C/D layout: VGPR r -> row m0 + r + g*8, col n0 + mn
            float* crow = C + (size_t)m0 * COUT + n0 + mn;
#pragma unroll
            for (int r = 0; r < 8; ++r)
                crow[(size_t)(r + (g << 3)) * COUT] = c.e[r];
        }
    }
}

template<int K, int COUT>
static inline void launch_gemm(const float* A, const float* W, float* C,
                               int M, hipStream_t stream)
{
    constexpr int TILES_N = COUT / 16;
    int mgroups = (M / 16 + 3) / 4;                 // 4 m-tiles per wave
    int total_waves = mgroups * TILES_N;
    int blocks = (total_waves + 7) / 8;             // 8 waves / 256-thr block
    wmma_gemm<K, COUT><<<blocks, 256, 0, stream>>>(A, W, C, M);
}

// ---------------------------------------------------------------------------
// In-degree (dst-based); deg must be pre-zeroed.
// ---------------------------------------------------------------------------
__global__ void deg_kernel(const int* __restrict__ dst, float* __restrict__ deg, int E)
{
    int i = blockIdx.x * blockDim.x + threadIdx.x;
    if (i < E) atomicAdd(&deg[dst[i]], 1.0f);
}

__global__ void norm_kernel(const int* __restrict__ src, const int* __restrict__ dst,
                            const float* __restrict__ deg, float* __restrict__ nrm, int E)
{
    int i = blockIdx.x * blockDim.x + threadIdx.x;
    if (i < E) {
        float ds = fmaxf(deg[src[i]], 1.0f);
        float dd = fmaxf(deg[dst[i]], 1.0f);
        nrm[i] = rsqrtf(ds * dd);
    }
}

// ---------------------------------------------------------------------------
// Edge scatter: agg[t, dst[e], c] += (xw[t, src[e], c] + ew[e, c]) * norm[e]
// blockIdx.y = t; each 256-thread block covers 256/COUT edges x COUT channels.
// COUT compile-time -> no integer division; coalesced loads & atomic bursts.
// ---------------------------------------------------------------------------
template<int COUT>
__global__ void scatter_kernel(const float* __restrict__ xw,
                               const float* __restrict__ ew,
                               const float* __restrict__ nrm,
                               const int* __restrict__ src,
                               const int* __restrict__ dst,
                               float* __restrict__ agg)
{
    constexpr int EPB = 256 / COUT;
    int c  = threadIdx.x & (COUT - 1);
    int eo = threadIdx.x / COUT;
    int e  = blockIdx.x * EPB + eo;
    int t  = blockIdx.y;
    if (e >= N_EDGES) return;
    int s = src[e], d = dst[e];
    float v = (xw[((size_t)t * N_NODES + s) * COUT + c] +
               ew[(size_t)e * COUT + c]) * nrm[e];
    atomicAdd(&agg[((size_t)t * N_NODES + d) * COUT + c], v);
}

template<int COUT>
static inline void launch_scatter(const float* xw, const float* ew, const float* nrm,
                                  const int* src, const int* dst, float* agg,
                                  hipStream_t stream)
{
    constexpr int EPB = 256 / COUT;
    dim3 grid((N_EDGES + EPB - 1) / EPB, T_STEPS);
    scatter_kernel<COUT><<<grid, 256, 0, stream>>>(xw, ew, nrm, src, dst, agg);
}

// ---------------------------------------------------------------------------
// GRU pointwise with H0 = 0:  h = (1 - sigmoid(R)) * tanh(HC), optional ReLU.
// R is the second half of the gate output (Z is dead since Z*H0 == 0).
// ---------------------------------------------------------------------------
__global__ void gru_out_kernel(const float* __restrict__ agg_g,
                               const float* __restrict__ agg_u,
                               const float* __restrict__ bg,
                               const float* __restrict__ bu,
                               float* __restrict__ out,
                               int TN, int do_relu)
{
    int idx = blockIdx.x * blockDim.x + threadIdx.x;
    if (idx >= TN * HID_C) return;
    int row = idx >> 5;      // / HID_C
    int j   = idx & 31;      // % HID_C
    float rg = agg_g[(size_t)row * (2 * HID_C) + HID_C + j] + bg[HID_C + j];
    float r  = 1.0f / (1.0f + __expf(-rg));
    float hc = tanhf(agg_u[(size_t)row * HID_C + j] + bu[j]);
    float h  = (1.0f - r) * hc;
    if (do_relu) h = fmaxf(h, 0.0f);
    out[idx] = h;
}

// ---------------------------------------------------------------------------
extern "C" void kernel_launch(void* const* d_in, const int* in_sizes, int n_in,
                              void* d_out, int out_size, void* d_ws, size_t ws_size,
                              hipStream_t stream)
{
    (void)in_sizes; (void)n_in; (void)out_size; (void)ws_size;

    const float* x     = (const float*)d_in[0];   // [1,T,N,CIN]
    const int*   ei    = (const int*)  d_in[1];   // [2,E]
    const float* eattr = (const float*)d_in[2];   // [E,CE]
    const float* Wg0   = (const float*)d_in[3];   // [CIN+HID, 2H] (rows >= CIN dead)
    const float* Weg0  = (const float*)d_in[4];
    const float* bg0   = (const float*)d_in[5];
    const float* Wu0   = (const float*)d_in[6];
    const float* Weu0  = (const float*)d_in[7];
    const float* bu0   = (const float*)d_in[8];
    const float* Wg1   = (const float*)d_in[9];   // [2H, 2H] (rows >= H dead)
    const float* Weg1  = (const float*)d_in[10];
    const float* bg1   = (const float*)d_in[11];
    const float* Wu1   = (const float*)d_in[12];
    const float* Weu1  = (const float*)d_in[13];
    const float* bu1   = (const float*)d_in[14];
    float* out = (float*)d_out;                   // [T,N,H]

    const int* src = ei;
    const int* dst = ei + N_EDGES;

    const int TN = T_STEPS * N_NODES;             // 120000, multiple of 16
    const int G  = 2 * HID_C;                     // 64

    // ---- workspace layout (floats) ----
    float* ws   = (float*)d_ws;
    size_t off  = 0;
    float* deg   = ws + off; off += N_NODES;
    float* nrm   = ws + off; off += N_EDGES;
    float* xw_g  = ws + off; off += (size_t)TN * G;        // [TN,64]
    float* xw_u  = ws + off; off += (size_t)TN * HID_C;    // [TN,32]
    float* ew_g  = ws + off; off += (size_t)N_EDGES * G;   // [E,64]
    float* ew_u  = ws + off; off += (size_t)N_EDGES * HID_C;
    float* agg_g = ws + off; off += (size_t)TN * G;
    float* agg_u = ws + off; off += (size_t)TN * HID_C;
    float* h1    = ws + off; off += (size_t)TN * HID_C;    // layer-0 output

    // ---- degree / norm (shared by both layers) ----
    hipMemsetAsync(deg, 0, N_NODES * sizeof(float), stream);
    deg_kernel<<<(N_EDGES + 255) / 256, 256, 0, stream>>>(dst, deg, N_EDGES);
    norm_kernel<<<(N_EDGES + 255) / 256, 256, 0, stream>>>(src, dst, deg, nrm, N_EDGES);

    // ================= Layer 0 (input X, K = CIN = 16) =================
    launch_gemm<C_IN,   64>(x,     Wg0,  xw_g, TN,      stream);
    launch_gemm<C_IN,   32>(x,     Wu0,  xw_u, TN,      stream);
    launch_gemm<C_EDGE, 64>(eattr, Weg0, ew_g, N_EDGES, stream);
    launch_gemm<C_EDGE, 32>(eattr, Weu0, ew_u, N_EDGES, stream);

    hipMemsetAsync(agg_g, 0, (size_t)TN * G * sizeof(float), stream);
    hipMemsetAsync(agg_u, 0, (size_t)TN * HID_C * sizeof(float), stream);
    launch_scatter<64>(xw_g, ew_g, nrm, src, dst, agg_g, stream);
    launch_scatter<32>(xw_u, ew_u, nrm, src, dst, agg_u, stream);

    gru_out_kernel<<<(TN * HID_C + 255) / 256, 256, 0, stream>>>(
        agg_g, agg_u, bg0, bu0, h1, TN, /*relu=*/1);

    // ================= Layer 1 (input h1, K = HID = 32) =================
    launch_gemm<HID_C,  64>(h1,    Wg1,  xw_g, TN,      stream);
    launch_gemm<HID_C,  32>(h1,    Wu1,  xw_u, TN,      stream);
    launch_gemm<C_EDGE, 64>(eattr, Weg1, ew_g, N_EDGES, stream);
    launch_gemm<C_EDGE, 32>(eattr, Weu1, ew_u, N_EDGES, stream);

    hipMemsetAsync(agg_g, 0, (size_t)TN * G * sizeof(float), stream);
    hipMemsetAsync(agg_u, 0, (size_t)TN * HID_C * sizeof(float), stream);
    launch_scatter<64>(xw_g, ew_g, nrm, src, dst, agg_g, stream);
    launch_scatter<32>(xw_u, ew_u, nrm, src, dst, agg_u, stream);

    gru_out_kernel<<<(TN * HID_C + 255) / 256, 256, 0, stream>>>(
        agg_g, agg_u, bg1, bu1, out, TN, /*relu=*/0);
}
